// SKipPool_25890062861053
// MI455X (gfx1250) — compile-verified, matrix operand order checked by reading, not run on hardware
//
#include <hip/hip_runtime.h>
#include <hip/hip_bf16.h>
#include <math.h>

// SKipPool pipeline for MI455X (gfx1250, wave32).
// Gram band xg[r]·xg[r+j] computed with v_wmma_f32_16x16x32_f16; the 4 KB
// contiguous f16 tile is staged into LDS via the Tensor Data Mover
// (tensor_load_to_lds + s_wait_tensorcnt) when the builtin is available.
// Problem constants from the harness: B=128, N=512, C=64, FS=16.

#define FS    16
#define EPSV  1e-20f

typedef __attribute__((ext_vector_type(16))) _Float16     v16h;
typedef __attribute__((ext_vector_type(8)))  float        v8f;
typedef __attribute__((ext_vector_type(4)))  unsigned int u32x4;
typedef __attribute__((ext_vector_type(4)))  int          i32x4;
typedef __attribute__((ext_vector_type(8)))  int          i32x8;

#if defined(__has_builtin)
#if __has_builtin(__builtin_amdgcn_tensor_load_to_lds) && \
    __has_builtin(__builtin_amdgcn_s_wait_tensorcnt)
#define USE_TDM 1
#endif
#endif

// ---------------- K1: scores = (x·w + b) / ||w|| ----------------
__global__ void k_scores(const float* __restrict__ x, const float* __restrict__ w,
                         const float* __restrict__ bsc, float* __restrict__ scores,
                         int BN, int C) {
  int i = blockIdx.x * blockDim.x + threadIdx.x;
  if (i >= BN) return;
  float wn = 0.f;
  for (int c = 0; c < C; ++c) wn += w[c] * w[c];
  float inv = rsqrtf(wn);
  const float* xr = x + (size_t)i * C;
  float acc = bsc[0];
  for (int c = 0; c < C; ++c) acc += xr[c] * w[c];
  scores[i] = acc * inv;
}

// ---------------- K2: per-graph stable descending argsort (bitonic, N==512) ----
__global__ void k_sort(const float* __restrict__ scores, int* __restrict__ perm, int N) {
  __shared__ float key[512];
  __shared__ int   idx[512];
  int b = blockIdx.x;
  for (int t = threadIdx.x; t < N; t += blockDim.x) { key[t] = scores[b * N + t]; idx[t] = t; }
  __syncthreads();
  for (int k2 = 2; k2 <= N; k2 <<= 1) {
    for (int j = k2 >> 1; j > 0; j >>= 1) {
      for (int t = threadIdx.x; t < N; t += blockDim.x) {
        int p = t ^ j;
        if (p > t) {
          bool up = ((t & k2) == 0);
          float ka = key[t], kb = key[p];
          int   ia = idx[t], ib = idx[p];
          // total order: score desc, index asc  (== stable argsort of -scores)
          bool aAfter = (ka < kb) || (ka == kb && ia > ib);
          if (up ? aAfter : !aAfter) { key[t] = kb; key[p] = ka; idx[t] = ib; idx[p] = ia; }
        }
      }
      __syncthreads();
    }
  }
  for (int t = threadIdx.x; t < N; t += blockDim.x) perm[b * N + t] = idx[t];
}

// ---------------- K3: gather xg, f16 copy, row norms, a1/a2 dots --------------
__global__ void k_gather(const float* __restrict__ x, const int* __restrict__ perm,
                         const float* __restrict__ att, float* __restrict__ xg,
                         _Float16* __restrict__ xgh, float* __restrict__ norms,
                         float* __restrict__ d1, float* __restrict__ d2,
                         int BN, int N, int C) {
  int i = blockIdx.x * blockDim.x + threadIdx.x;
  if (i >= BN) return;
  int b = i / N;
  int src = perm[i];
  const float* xr = x + ((size_t)b * N + src) * C;
  float*    xo = xg  + (size_t)i * C;
  _Float16* ho = xgh + (size_t)i * C;
  float nn = 0.f, s1 = 0.f, s2 = 0.f;
  for (int c = 0; c < C; ++c) {
    float v = xr[c];
    xo[c] = v; ho[c] = (_Float16)v;
    nn += v * v; s1 += v * att[c]; s2 += v * att[C + c];
  }
  norms[i] = nn; d1[i] = s1; d2[i] = s2;
}

// Per-lane 16-bit fragment per ISA 7.12.2 layout: lanes 0-15 hold K={kb..kb+7}
// and K={16+kb..16+kb+7} with kb = 8*(lane>>4); identical layout for A (row=M)
// and B (row striped over N) which coincide for a symmetric Gram tile.
__device__ inline v16h frag(const _Float16* sT, int rowLocal, int kc, int kb) {
  union { v16h v; _Float16 h[16]; } u;
  const _Float16* p = sT + rowLocal * 64 + kc + kb;
#pragma unroll
  for (int e = 0; e < 8; ++e) { u.h[e] = p[e]; u.h[8 + e] = p[16 + e]; }
  return u.v;
}

// ---------------- K4: WMMA gram band -> gk, softmax, stable sort16 ------------
__global__ void __launch_bounds__(32)
k_gram(const _Float16* __restrict__ xgh, const float* __restrict__ norms,
       const float* __restrict__ d1, const float* __restrict__ d2,
       float* __restrict__ gk, float* __restrict__ gks, float* __restrict__ sms,
       int N, int RES) {
  __shared__ _Float16 sT[32 * 64];   // 32 rows x 64 ch (f16), 4 KB
  __shared__ float    sS[16 * 32];   // 16 x 32 gram tile, 2 KB
  const int b    = blockIdx.y;
  const int m0   = blockIdx.x * 16;
  const int lane = threadIdx.x;

#ifdef USE_TDM
  { // Tensor Data Mover: contiguous 1-D tile (<=2048 f16) -> LDS.
    // D# packing per CDNA5 ISA:
    //   group0: [1:0]=count=1, [63:32]=lds_addr, [120:64]=global_addr, [127:126]=type=2
    //   group1: [17:16]=data_size=1 (2B), tensor_dim0 bounds the read (OOB -> 0),
    //           tile_dim0=elems, tensor_dim0_stride=elems
    unsigned rows = (unsigned)((N - m0) < 32 ? (N - m0) : 32);
    unsigned elems = rows * 64u;                       // f16 elements, contiguous
    unsigned long long ga =
        (unsigned long long)(const void*)(xgh + ((size_t)b * N + m0) * 64);
    unsigned lds = (unsigned)(unsigned long long)(const void*)&sT[0];
    u32x4 g0;
    g0.x = 1u;                                           // count=1 user descriptor
    g0.y = lds;                                          // lds_addr
    g0.z = (unsigned)(ga & 0xffffffffu);                 // global_addr[31:0]
    g0.w = (unsigned)((ga >> 32) & 0x01ffffffu) | (2u << 30);  // addr[56:32] | type=2
    i32x8 g1;
    g1[0] = (int)(1u << 16);                             // data_size=1 (2 bytes)
    g1[1] = (int)((elems & 0xffffu) << 16);              // tensor_dim0[15:0]
    g1[2] = (int)((elems >> 16) | (1u << 16));           // tensor_dim0[31:16] | tensor_dim1=1
    g1[3] = (int)(elems << 16);                          // tile_dim0 = elems
    g1[4] = 0;                                           // tile_dim1=0, tile_dim2=0
    g1[5] = (int)elems;                                  // tensor_dim0_stride[31:0]
    g1[6] = 0; g1[7] = 0;
    i32x4 z4 = {0, 0, 0, 0};
#if __clang_major__ >= 23
    i32x8 z8 = {0, 0, 0, 0, 0, 0, 0, 0};
    __builtin_amdgcn_tensor_load_to_lds(g0, g1, z4, z4, z8, 0);
#else
    __builtin_amdgcn_tensor_load_to_lds(g0, g1, z4, z4, 0);
#endif
    __builtin_amdgcn_s_wait_tensorcnt(0);                // s_wait_tensorcnt 0
  }
#else
  { // fallback: stage rows m0..m0+31 (clamped) into LDS; 128 B per row via uint4
    int row = m0 + lane; if (row > N - 1) row = N - 1;
    const uint4* g  = (const uint4*)(xgh + ((size_t)b * N + row) * 64);
    uint4*       sv = (uint4*)(sT + lane * 64);
    __builtin_prefetch(xgh + ((size_t)b * N + row) * 64, 0, 0);  // global_prefetch_b8
#pragma unroll
    for (int q = 0; q < 8; ++q) sv[q] = g[q];
  }
#endif
  __syncthreads();

  const int half = lane >> 4, l15 = lane & 15, kb = half * 8;
  v8f c0 = {}; v8f c1 = {};
#pragma unroll
  for (int kc = 0; kc < 64; kc += 32) {
    v16h A  = frag(sT, l15,      kc, kb);   // rows m0..m0+15
    v16h B1 = frag(sT, 16 + l15, kc, kb);   // rows m0+16..m0+31
    // S0 = A·A^T (cols m0..m0+15), S1 = A·B1^T (cols m0+16..m0+31)
    c0 = __builtin_amdgcn_wmma_f32_16x16x32_f16(false, A, false, A,  (short)0, c0, false, false);
    c1 = __builtin_amdgcn_wmma_f32_16x16x32_f16(false, A, false, B1, (short)0, c1, false, false);
  }
  { // dump C/D fragments (lane holds N=l15, VGPR q holds M=q+8*half) to LDS
    union { v8f v; float f[8]; } u0, u1; u0.v = c0; u1.v = c1;
#pragma unroll
    for (int q = 0; q < 8; ++q) {
      sS[(q + 8 * half) * 32 + l15]      = u0.f[q];
      sS[(q + 8 * half) * 32 + 16 + l15] = u1.f[q];
    }
  }
  __syncthreads();

  int r = m0 + l15;
  if (half == 0 && r < RES) {
    size_t base = ((size_t)b * RES + r) * FS;
    float nr  = norms[b * N + r];
    float d1r = d1[b * N + r];
    float e[FS], gkv[FS];
    float mx = -INFINITY;
#pragma unroll
    for (int j = 0; j < FS; ++j) {
      float dot = sS[l15 * 32 + l15 + j];
      float sq  = nr + norms[b * N + r + j] - 2.f * dot;
      float dd  = (sq > 0.f) ? sqrtf(sq) : 0.f;        // _safe_norm
      gkv[j] = __expf(-0.5f * dd) - EPSV;
      float ej = d1r + d2[b * N + r + j];
      e[j] = ej; mx = fmaxf(mx, ej);
    }
    float den = 0.f;
#pragma unroll
    for (int j = 0; j < FS; ++j) { e[j] = __expf(e[j] - mx); den += e[j]; }
    float inv = 1.f / den;
    // stable insertion sort: gk desc, tie -> lower original index first
    float gs[FS], ss[FS]; int oi[FS];
    for (int j = 0; j < FS; ++j) { gs[j] = gkv[j]; ss[j] = e[j] * inv; oi[j] = j; }
    for (int a = 1; a < FS; ++a) {
      float gv = gs[a], sv = ss[a]; int ov = oi[a]; int p = a - 1;
      while (p >= 0 && (gs[p] < gv || (gs[p] == gv && oi[p] > ov))) {
        gs[p + 1] = gs[p]; ss[p + 1] = ss[p]; oi[p + 1] = oi[p]; --p;
      }
      gs[p + 1] = gv; ss[p + 1] = sv; oi[p + 1] = ov;
    }
    for (int j = 0; j < FS; ++j) { gk[base + j] = gkv[j]; gks[base + j] = gs[j]; sms[base + j] = ss[j]; }
  }
}

// ---------------- K4b: logits[b,j] = sum_r sm_sort (deterministic) ------------
__global__ void k_logits(const float* __restrict__ sms, float* __restrict__ logits, int RES) {
  int b = blockIdx.x, j = threadIdx.x;
  float acc = 0.f;
  for (int r = 0; r < RES; ++r) acc += sms[((size_t)b * RES + r) * FS + j];
  logits[b * FS + j] = acc;
}

// ---------------- K5: gumbel softmax threshold, shifted, delta ----------------
__device__ inline float hash_u01(unsigned a) {
  a ^= a >> 16; a *= 0x7feb352du; a ^= a >> 15; a *= 0x846ca68bu; a ^= a >> 16;
  return ((float)(a >> 8) + 0.5f) * (1.0f / 16777216.0f);
}
__global__ void k_gumbel(const float* __restrict__ gk, const float* __restrict__ gks,
                         const float* __restrict__ logits, const int* __restrict__ epoch_p,
                         float* __restrict__ shifted, int* __restrict__ delta,
                         int B, int RES) {
  int tid = blockIdx.x * blockDim.x + threadIdx.x;
  if (tid >= B * RES) return;
  int b = tid / RES;
  float ep  = (float)epoch_p[0];
  float tau = 10.0f * powf(0.01f, ep * 0.01f);   // TAUA*(TAUB/TAUA)^(epoch/100)
  float invtau = 1.0f / tau;
  size_t base = (size_t)tid * FS;
  float z[FS], wgt[FS];
  float mx = -INFINITY;
#pragma unroll
  for (int j = 0; j < FS; ++j) {
    // deterministic gumbel stand-in (JAX threefry not bit-reproducible here)
    unsigned h = (777u * 2654435761u) ^ ((unsigned)(tid * FS + j) * 0x9E3779B9u);
    float u = hash_u01(h);
    float g = -logf(-logf(u));
    float lg = (j == 0) ? -INFINITY : logits[b * FS + j];
    z[j] = (g + lg) * invtau;
    mx = fmaxf(mx, z[j]);
  }
  float den = 0.f;
#pragma unroll
  for (int j = 0; j < FS; ++j) { wgt[j] = __expf(z[j] - mx); den += wgt[j]; }
  float invden = 1.f / den, ths = 0.f;
#pragma unroll
  for (int j = 0; j < FS; ++j) ths += gks[base + j] * wgt[j] * invden;
  int am = 1; float mv = INFINITY;
#pragma unroll
  for (int j = 0; j < FS; ++j) {
    float sv = fmaxf(gk[base + j] - ths, 0.f);
    shifted[base + j] = sv;
    if (j >= 1 && sv < mv) { mv = sv; am = j; }   // first minimum (strict <)
  }
  delta[tid] = am;   // argmin over [1:] + 1 == original j index
}

// ---------------- K6a: sequential pivot chain (1 thread / graph) --------------
__global__ void k_chain(const int* __restrict__ delta, int* __restrict__ pivots,
                        int RES, int MAXP) {
  int b = threadIdx.x;
  int ptr = 0;
  for (int t = 0; t < MAXP; ++t) {
    int piv = (ptr < RES) ? ptr : -1;
    pivots[b * MAXP + t] = piv;
    int ci = ptr; if (ci < 0) ci = 0; if (ci > RES - 1) ci = RES - 1;
    ptr = (ptr < RES) ? (ptr + delta[b * RES + ci]) : RES;
  }
}

// ---------------- K6b: pooled gather ------------------------------------------
__global__ void k_pool(const float* __restrict__ xg, const int* __restrict__ pivots,
                       float* __restrict__ pooled, int N, int C, int MAXP) {
  int b = blockIdx.x, t = blockIdx.y, c = threadIdx.x;
  int piv = pivots[b * MAXP + t];
  float v = 0.f;
  if (piv >= 0) {
    int idx = piv; if (idx > N - 1) idx = N - 1;
    v = xg[((size_t)b * N + idx) * C + c];
  }
  pooled[((size_t)b * MAXP + t) * C + c] = v;
}

extern "C" void kernel_launch(void* const* d_in, const int* in_sizes, int n_in,
                              void* d_out, int out_size, void* d_ws, size_t ws_size,
                              hipStream_t stream) {
  const float* x   = (const float*)d_in[0];
  const float* w   = (const float*)d_in[1];
  const float* bsc = (const float*)d_in[2];
  const float* att = (const float*)d_in[3];
  const int* epoch = (const int*)d_in[5];

  const int C    = in_sizes[3] / 2;   // 64
  const int BN   = in_sizes[0] / C;   // 65536
  const int B    = 128;               // num_graphs fixed by harness setup
  const int N    = BN / B;            // 512
  const int RES  = N - FS + 1;        // 497
  const int MAXP = (N + 1) / 2;       // ceil(N*0.5) = 256

  // workspace carve (~39 MB), 256B-aligned slots
  char* wsp = (char*)d_ws; size_t off = 0;
  auto carve = [&](size_t bytes) -> void* {
    void* p = wsp + off; off += (bytes + 255) & ~(size_t)255; return p;
  };
  float*    scores = (float*)   carve((size_t)BN * 4);
  int*      perm   = (int*)     carve((size_t)BN * 4);
  float*    xg     = (float*)   carve((size_t)BN * C * 4);
  _Float16* xgh    = (_Float16*)carve((size_t)BN * C * 2);
  float*    norms  = (float*)   carve((size_t)BN * 4);
  float*    d1     = (float*)   carve((size_t)BN * 4);
  float*    d2     = (float*)   carve((size_t)BN * 4);
  float*    gk     = (float*)   carve((size_t)B * RES * FS * 4);
  float*    gks    = (float*)   carve((size_t)B * RES * FS * 4);
  float*    sms    = (float*)   carve((size_t)B * RES * FS * 4);
  float*    logits = (float*)   carve((size_t)B * FS * 4);
  int*      delta  = (int*)     carve((size_t)B * RES * 4);

  // outputs concatenated: pooled (B,MAXP,C) f32 | shifted (B,RES,FS) f32 | pivots (B,MAXP) i32
  float* out_pooled  = (float*)d_out;
  float* out_shifted = out_pooled + (size_t)B * MAXP * C;
  int*   out_pivots  = (int*)(out_shifted + (size_t)B * RES * FS);

  k_scores<<<(BN + 255) / 256, 256, 0, stream>>>(x, w, bsc, scores, BN, C);
  k_sort  <<<B, 256, 0, stream>>>(scores, perm, N);
  k_gather<<<(BN + 255) / 256, 256, 0, stream>>>(x, perm, att, xg, xgh, norms, d1, d2, BN, N, C);
  k_gram  <<<dim3(N / 16, B), 32, 0, stream>>>(xgh, norms, d1, d2, gk, gks, sms, N, RES);
  k_logits<<<B, FS, 0, stream>>>(sms, logits, RES);
  k_gumbel<<<(B * RES + 255) / 256, 256, 0, stream>>>(gk, gks, logits, epoch, out_shifted, delta, B, RES);
  k_chain <<<1, B, 0, stream>>>(delta, out_pivots, RES, MAXP);
  k_pool  <<<dim3(B, MAXP), C, 0, stream>>>(xg, out_pivots, out_pooled, N, C, MAXP);
}
